// Discriminator_61907658604900
// MI455X (gfx1250) — matrix operand order repaired and checked by reading, not compiled, not run on hardware
//
#include <hip/hip_runtime.h>
#include <math.h>

typedef __attribute__((ext_vector_type(16))) _Float16 v16h;
typedef __attribute__((ext_vector_type(8)))  float    v8f;

#define B_    64
#define L_    128
#define E_    300
#define EP_   304      // padded row stride (16B-aligned rows, pad folded into GEMM K)
#define FTOT  1000

// ---------------- embedding gather + f16 convert into padded-stride layout ---------
// embh[b*L + l][0..303]: cols 0..299 = emb_w[x[b,l]], cols 300..303 = 0; +64 zero tail
__global__ void k_embed(const int* __restrict__ x, const float* __restrict__ embw,
                        _Float16* __restrict__ embh, int total, int padTotal) {
  int idx = blockIdx.x * blockDim.x + threadIdx.x;
  if (idx >= padTotal) return;
  if (idx >= total) { embh[idx] = (_Float16)0.f; return; }
  int row = idx / EP_;
  int e   = idx - row * EP_;
  float v = 0.f;
  if (e < E_) {
    int tok = x[row];
    v = embw[(long long)tok * E_ + e];
  }
  embh[idx] = (_Float16)v;
}

// ---------------- weight repack into WMMA B-fragment layout (zero padded) ----------
// GEMM K index = tap*EP_ + e.  element (tile = nt*KT + kt, lane, i):
//   n = nt*16 + (lane&15); K = kt*32 + (lane>>4)*16 + i   (contiguous k order, same as the A load)
// NT4 includes zero-padded N-tiles so the GEMM K-loop is branch-free.
__global__ void k_packw(const float* __restrict__ w, _Float16* __restrict__ wp,
                        int KT, int NT4, int f, int nf) {
  int idx = blockIdx.x * blockDim.x + threadIdx.x;
  int totalEl = KT * NT4 * 512;
  if (idx >= totalEl) return;
  int i    = idx & 15;
  int lane = (idx >> 4) & 31;
  int tile = idx >> 9;
  int kt   = tile % KT;
  int nt   = tile / KT;
  int n   = nt * 16 + (lane & 15);
  int K   = kt * 32 + (lane >> 4) * 16 + i;
  int tap = K / EP_;
  int e   = K - tap * EP_;
  float v = 0.f;
  if (tap < f && e < E_ && n < nf) v = w[(size_t)(tap * E_ + e) * nf + n];
  wp[idx] = (_Float16)v;
}

__global__ void k_zero(float* __restrict__ p, int n) {
  int i = blockIdx.x * blockDim.x + threadIdx.x;
  if (i < n) p[i] = 0.f;
}

// ---------------- conv-as-GEMM with WMMA, fused bias+ReLU+max-pool -----------------
// 4 waves/block; each wave computes a 16-row x 64-col D tile.
// Branch-free, ping/pong double-buffered K loop (partial loadcnt waits).
__global__ void __launch_bounds__(128)
k_conv_gemm(const _Float16* __restrict__ emb, const _Float16* __restrict__ wp,
            const float* __restrict__ bias, float* __restrict__ pooled,
            int Lout, int nf, int KT, int colOff) {
  const int wave   = threadIdx.x >> 5;
  const int lane   = threadIdx.x & 31;
  const int mtile  = blockIdx.x * 4 + wave;
  const int ntBase = blockIdx.y * 4;
  const int Mtot   = B_ * Lout;
  const int kh     = lane >> 4;        // K-half select for A fragment

  // A row = contiguous padded conv window: emb + (b*L + t)*EP_, 16B aligned
  int m  = mtile * 16 + (lane & 15);
  int mc = (m < Mtot) ? m : (Mtot - 1);
  int b  = mc / Lout;
  int t  = mc - b * Lout;
  const uint4* __restrict__ rowp =
      (const uint4*)(emb + (size_t)(b * L_ + t) * EP_);

  union AB { v16h h; uint4 u[2]; };
  const size_t bstride = (size_t)KT * 512;  // halfs per N-tile column of wp
  const _Float16* __restrict__ wpBase = wp + (size_t)ntBase * bstride;

  auto loadA = [&](int kt, AB& d) {
    const int a0 = kt * 4 + kh * 2;         // uint4 index within row
    d.u[0] = rowp[a0];
    d.u[1] = rowp[a0 + 1];
  };
  auto loadB = [&](int kt, AB* d) {
#pragma unroll
    for (int j = 0; j < 4; ++j) {
      const uint4* __restrict__ tp =
          (const uint4*)(wpBase + (size_t)j * bstride + (size_t)kt * 512);
      d[j].u[0] = tp[lane * 2];
      d[j].u[1] = tp[lane * 2 + 1];
    }
  };

  v8f acc[4] = {};
  AB Ap, Bp[4], Aq, Bq[4];

  loadA(0, Ap);
  loadB(0, Bp);
  int kt = 0;
  for (; kt + 2 <= KT; kt += 2) {
    // prefetch kt+1 into pong while computing ping
    loadA(kt + 1, Aq);
    loadB(kt + 1, Bq);
#pragma unroll
    for (int j = 0; j < 4; ++j)
      acc[j] = __builtin_amdgcn_wmma_f32_16x16x32_f16(
          false, Ap.h, false, Bp[j].h, (short)0, acc[j], false, false);
    // prefetch kt+2 into ping (kt+2==KT -> dead loads into mapped pad) …
    loadA(kt + 2, Ap);
    loadB(kt + 2, Bp);
#pragma unroll
    for (int j = 0; j < 4; ++j)
      acc[j] = __builtin_amdgcn_wmma_f32_16x16x32_f16(
          false, Aq.h, false, Bq[j].h, (short)0, acc[j], false, false);
  }
  if (kt < KT) {   // KT odd: last fragment already resident in ping
#pragma unroll
    for (int j = 0; j < 4; ++j)
      acc[j] = __builtin_amdgcn_wmma_f32_16x16x32_f16(
          false, Ap.h, false, Bp[j].h, (short)0, acc[j], false, false);
  }

  // Epilogue: C/D layout lane&15 = N, VGPR r -> M = (lane>=16?8:0)+r.
  // bias + relu, run-max over consecutive rows sharing b, one atomic per run.
#pragma unroll
  for (int j = 0; j < 4; ++j) {
    int col = (ntBase + j) * 16 + (lane & 15);
    if (col >= nf) continue;             // also discards zero-padded N-tiles
    float bv = bias[col];
    int   curb = -1;
    float runm = 0.f;
#pragma unroll
    for (int r = 0; r < 8; ++r) {
      int mm = mtile * 16 + kh * 8 + r;
      if (mm >= Mtot) continue;
      int bb = mm / Lout;
      float v = fmaxf(acc[j][r] + bv, 0.f);
      if (bb != curb) {
        if (curb >= 0)
          atomicMax((int*)&pooled[(size_t)curb * FTOT + colOff + col],
                    __float_as_int(runm));
        curb = bb; runm = v;
      } else {
        runm = fmaxf(runm, v);
      }
    }
    if (curb >= 0)
      atomicMax((int*)&pooled[(size_t)curb * FTOT + colOff + col],
                __float_as_int(runm));
  }
}

// ---------------- highway + linear + log_softmax (tiny, f32) -----------------------
__global__ void __launch_bounds__(256)
k_highway(const float* __restrict__ pooled, const float* __restrict__ hw_w,
          const float* __restrict__ hw_b, const float* __restrict__ hwr_w,
          const float* __restrict__ hwr_b, const float* __restrict__ lin_w,
          const float* __restrict__ lin_b, float* __restrict__ pred) {
  const int b      = blockIdx.x;
  const int branch = blockIdx.y;
  const float* W  = branch ? hwr_w : hw_w;
  const float* Wb = branch ? hwr_b : hw_b;
  __shared__ float ps[FTOT];
  __shared__ float hs[FTOT];
  __shared__ float red0[256];
  __shared__ float red1[256];
  const int tid = threadIdx.x;
  const float* prow = pooled + (size_t)(branch * B_ + b) * FTOT;
  for (int j = tid; j < FTOT; j += 256) ps[j] = prow[j];
  __syncthreads();
  for (int j = tid; j < FTOT; j += 256) {
    const float* wr = W + (size_t)j * FTOT;
    float h = Wb[j];
    for (int k = 0; k < FTOT; ++k) h = fmaf(ps[k], wr[k], h);
    float s = 1.f / (1.f + expf(-h));
    hs[j] = s * fmaxf(h, 0.f) + (1.f - s) * ps[j];
  }
  __syncthreads();
  float p0 = 0.f, p1 = 0.f;
  for (int j = tid; j < FTOT; j += 256) {
    p0 = fmaf(hs[j], lin_w[j], p0);
    p1 = fmaf(hs[j], lin_w[FTOT + j], p1);
  }
  red0[tid] = p0; red1[tid] = p1;
  __syncthreads();
  for (int s = 128; s > 0; s >>= 1) {
    if (tid < s) { red0[tid] += red0[tid + s]; red1[tid] += red1[tid + s]; }
    __syncthreads();
  }
  if (tid == 0) {
    float l0 = red0[0] + lin_b[0];
    float l1 = red1[0] + lin_b[1];
    float mx = fmaxf(l0, l1);
    float lse = mx + logf(expf(l0 - mx) + expf(l1 - mx));
    float* pr = pred + (size_t)(branch * B_ + b) * 2;
    pr[0] = l0 - lse;
    pr[1] = l1 - lse;
  }
}

// ---------------- cosine-similarity head ------------------------------------------
// values[i] = (sum_j rn_j) . pn_i  since cos is bilinear in normalized vectors
__global__ void __launch_bounds__(64)
k_cosine(const float* __restrict__ pred, float* __restrict__ out) {
  __shared__ float Rx, Ry, total;
  __shared__ float vals[B_];
  const int i = threadIdx.x;
  if (i == 0) {
    float rx = 0.f, ry = 0.f;
    for (int j = 0; j < 16; ++j) {
      float x = pred[(B_ + j) * 2 + 0];
      float y = pred[(B_ + j) * 2 + 1];
      float n = fmaxf(sqrtf(x * x + y * y), 1e-8f);
      rx += x / n; ry += y / n;
    }
    Rx = rx; Ry = ry;
  }
  __syncthreads();
  float x = pred[i * 2], y = pred[i * 2 + 1];
  float n = fmaxf(sqrtf(x * x + y * y), 1e-8f);
  vals[i] = (Rx * x + Ry * y) / n;
  __syncthreads();
  if (i == 0) {
    float tsum = 0.f;
    for (int j = 0; j < B_; ++j) tsum += vals[j];
    total = tsum;
  }
  __syncthreads();
  out[i] = logf(vals[i] / total);
}

extern "C" void kernel_launch(void* const* d_in, const int* in_sizes, int n_in,
                              void* d_out, int out_size, void* d_ws, size_t ws_size,
                              hipStream_t stream) {
  (void)in_sizes; (void)n_in; (void)out_size; (void)ws_size;
  const int*   x    = (const int*)d_in[0];
  const float* embw = (const float*)d_in[1];
  const float* cw[2][3];
  const float* cb[2][3];
  int p = 2;   // dict order: conv_w{f}, conv_b{f}, convr_w{f}, convr_b{f} for f=3,4,5
  for (int f = 0; f < 3; ++f) {
    cw[0][f] = (const float*)d_in[p++];
    cb[0][f] = (const float*)d_in[p++];
    cw[1][f] = (const float*)d_in[p++];
    cb[1][f] = (const float*)d_in[p++];
  }
  const float* hw_w  = (const float*)d_in[14];
  const float* hw_b  = (const float*)d_in[15];
  const float* hwr_w = (const float*)d_in[16];
  const float* hwr_b = (const float*)d_in[17];
  const float* lin_w = (const float*)d_in[18];
  const float* lin_b = (const float*)d_in[19];

  const int FS[3] = {3, 4, 5};
  const int NF[3] = {300, 300, 400};

  // workspace carve-out (all offsets 1 KiB aligned -> 16B aligned b128 loads)
  char* ws = (char*)d_ws;
  size_t cur = 0;
  auto alloc = [&](size_t bytes) -> void* {
    void* r = ws + cur;
    cur = (cur + bytes + 1023) & ~((size_t)1023);
    return r;
  };

  const int embTotal = B_ * L_ * EP_;
  const int embPad   = embTotal + 64;              // A K-tail overread cover (zeroed)
  _Float16* embh = (_Float16*)alloc((size_t)embPad * sizeof(_Float16));

  _Float16* wp[2][3];
  int KT[3], NT4[3];
  for (int f = 0; f < 3; ++f) {
    int Ktot = FS[f] * EP_;                        // padded K (stride EP_ per tap)
    KT[f]  = (Ktot + 31) / 32;
    NT4[f] = ((NF[f] + 15) / 16 + 3) & ~3;         // pad N-tiles to multiple of 4
    size_t sz = (size_t)KT[f] * NT4[f] * 512 * sizeof(_Float16);
    wp[0][f] = (_Float16*)alloc(sz + 1024);        // +1 tile: pipeline dead-load pad
  }
  // second branch buffers after first (keeps every wp overread inside ws)
  for (int f = 0; f < 3; ++f) {
    size_t sz = (size_t)KT[f] * NT4[f] * 512 * sizeof(_Float16);
    wp[1][f] = (_Float16*)alloc(sz + 1024);
  }
  float* pooled = (float*)alloc((size_t)2 * B_ * FTOT * sizeof(float));
  float* pred   = (float*)alloc((size_t)2 * B_ * 2 * sizeof(float));

  { int n = 2 * B_ * FTOT; k_zero<<<(n + 255) / 256, 256, 0, stream>>>(pooled, n); }
  k_embed<<<(embPad + 255) / 256, 256, 0, stream>>>(x, embw, embh, embTotal, embPad);

  for (int br = 0; br < 2; ++br)
    for (int f = 0; f < 3; ++f) {
      int tot = KT[f] * NT4[f] * 512;
      k_packw<<<(tot + 255) / 256, 256, 0, stream>>>(cw[br][f], wp[br][f],
                                                     KT[f], NT4[f], FS[f], NF[f]);
    }

  int colOff = 0;
  for (int f = 0; f < 3; ++f) {
    int Lout   = L_ - FS[f] + 1;
    int Mtiles = (B_ * Lout + 15) / 16;            // 504 / 500 / 496, all %4 == 0
    int NB     = NT4[f] / 4;
    for (int br = 0; br < 2; ++br) {
      k_conv_gemm<<<dim3(Mtiles / 4, NB), 128, 0, stream>>>(
          embh, wp[br][f], cb[br][f], pooled + (size_t)br * B_ * FTOT,
          Lout, NF[f], KT[f], colOff);
    }
    colOff += NF[f];
  }

  k_highway<<<dim3(B_, 2), 256, 0, stream>>>(pooled, hw_w, hw_b, hwr_w, hwr_b,
                                             lin_w, lin_b, pred);
  k_cosine<<<1, 64, 0, stream>>>(pred, (float*)d_out);
}
